// RPMNetEarlyFusion_33406255628957
// MI455X (gfx1250) — matrix operand order repaired and checked by reading, not compiled, not run on hardware
//
#include <hip/hip_runtime.h>
#include <cstdint>

// ---------------------------------------------------------------------------
// RPMNet early-fusion for MI455X (gfx1250, wave32).
// All matmuls via v_wmma_f32_16x16x32_bf16 (f32 accumulate).
// A fragments held in registers and reused across all cout tiles (3-6x HBM
// read reduction); pre-activation f32 tensor streamed with non-temporal
// hints so the bf16 fragment buffers stay L2-resident.
// ---------------------------------------------------------------------------

#define BATCH 8
#define NPTS  2048
#define KNBR  64
#define RAD2  0.09f

static constexpr int    M_CONV     = BATCH * NPTS * KNBR;   // 1,048,576 fused rows
static constexpr int    TILES_CONV = M_CONV / 16;           // 65,536
static constexpr int    M_PT       = BATCH * NPTS;          // 16,384 pooled rows
static constexpr int    TILES_PT   = M_PT / 16;             // 1,024

typedef __attribute__((ext_vector_type(16))) __bf16 v16bf;
typedef __attribute__((ext_vector_type(8)))  float  v8f;

union FragAB {
  uint4        q[2];
  unsigned int u[8];
  v16bf        v;
};

static __device__ __forceinline__ unsigned int pack_bf2(float a, float b) {
  // round-to-nearest-even f32 -> bf16, pack pair (low = even K, high = odd K)
  unsigned int ua = __float_as_uint(a), ub = __float_as_uint(b);
  unsigned int ra = (ua + 0x7FFFu + ((ua >> 16) & 1u)) >> 16;
  unsigned int rb = (ub + 0x7FFFu + ((ub >> 16) & 1u)) >> 16;
  return (ra & 0xFFFFu) | (rb << 16);
}

// A-fragment K index for dword j given lane half (ISA 16-bit A 16x32 layout)
static __device__ __forceinline__ int a_pattern(int khalf, int j) {
  return (j < 4) ? (khalf + 2 * j) : (16 + khalf + 2 * (j - 4));
}

static __device__ __forceinline__ float angle3(float ax, float ay, float az,
                                               float bx, float by, float bz) {
  float cx = ay * bz - az * by;
  float cy = az * bx - ax * bz;
  float cz = ax * by - ay * bx;
  float cn = sqrtf(cx * cx + cy * cy + cz * cz);
  float dt = ax * bx + ay * by + az * bz;
  return atan2f(cn, dt);
}

// ---------------------------------------------------------------------------
// 1) Neighbor search: one wave per query point; ballot+ctz keeps ascending
//    index order == reference's sort-of-masked-indices semantics.
// ---------------------------------------------------------------------------
__global__ void knn_kernel(const float* __restrict__ xyz, int* __restrict__ gidx) {
  int wave = blockIdx.x * (blockDim.x >> 5) + (threadIdx.x >> 5);
  int lane = threadIdx.x & 31;
  int b = wave >> 11;            // /NPTS
  int n = wave & (NPTS - 1);
  const float* base = xyz + (size_t)b * NPTS * 3;
  float cx = base[n * 3 + 0], cy = base[n * 3 + 1], cz = base[n * 3 + 2];
  int* out = gidx + (size_t)wave * KNBR;
  int cnt = 0;
  for (int s = 0; s < NPTS; s += 32) {
    int m = s + lane;
    float dx = base[m * 3 + 0] - cx;
    float dy = base[m * 3 + 1] - cy;
    float dz = base[m * 3 + 2] - cz;
    float d2 = dx * dx + dy * dy + dz * dz;
    bool ok = (d2 <= RAD2) && (m != n);
    unsigned int mask = __builtin_amdgcn_ballot_w32(ok);
    while (mask != 0u && cnt < KNBR) {
      int l = __builtin_ctz(mask);
      mask &= mask - 1u;
      if (lane == 0) out[cnt] = s + l;
      ++cnt;
    }
    if (cnt >= KNBR) break;
  }
  if (lane == 0) {
    for (int c = cnt; c < KNBR; ++c) out[c] = n;  // pad with self (ref semantics)
  }
}

// ---------------------------------------------------------------------------
// 2) PPF feature build, written directly as layer-0 bf16 A-fragments
//    (10 channels zero-padded to K=32, 1 k-chunk).
// ---------------------------------------------------------------------------
__global__ void feat_kernel(const float* __restrict__ xyz, const float* __restrict__ nrm,
                            const int* __restrict__ gidx, unsigned int* __restrict__ dst) {
  int tid  = blockIdx.x * blockDim.x + threadIdx.x;     // TILES_CONV*32 threads
  int lane = tid & 31;
  int t    = tid >> 5;
  int row  = t * 16 + (lane & 15);
  int bn   = row / KNBR;
  int b    = bn / NPTS;
  int m    = gidx[row];
  const float* pc = xyz + (size_t)bn * 3;
  const float* nc = nrm + (size_t)bn * 3;
  const float* pm = xyz + ((size_t)b * NPTS + m) * 3;
  const float* nm = nrm + ((size_t)b * NPTS + m) * 3;
  float cx = pc[0], cy = pc[1], cz = pc[2];
  float dx = pm[0] - cx, dy = pm[1] - cy, dz = pm[2] - cz;
  float nrx = nc[0], nry = nc[1], nrz = nc[2];
  float nix = nm[0], niy = nm[1], niz = nm[2];
  float f[10];
  f[0] = cx; f[1] = cy; f[2] = cz;
  f[3] = dx; f[4] = dy; f[5] = dz;
  f[6] = angle3(nrx, nry, nrz, dx, dy, dz);
  f[7] = angle3(nix, niy, niz, dx, dy, dz);
  f[8] = angle3(nrx, nry, nrz, nix, niy, niz);
  f[9] = sqrtf(dx * dx + dy * dy + dz * dz);
  int khalf = (lane >> 4) * 8;
  unsigned int* d = dst + ((size_t)t * 32 + lane) * 8;  // kchunks = 1
#pragma unroll
  for (int j = 0; j < 8; ++j) {
    int kk = a_pattern(khalf, j);
    float v0 = (kk     < 10) ? f[kk]     : 0.f;
    float v1 = (kk + 1 < 10) ? f[kk + 1] : 0.f;
    d[j] = pack_bf2(v0, v1);
  }
}

// ---------------------------------------------------------------------------
// 3) Weight pre-pack into bf16 B-fragments (K x 16 per cout tile, K-chunked).
// ---------------------------------------------------------------------------
__global__ void pack_weight_kernel(const float* __restrict__ W, int Cin, int Cout,
                                   unsigned int* __restrict__ dst, int kchunks) {
  int lane  = threadIdx.x;               // blockDim = 32
  int c     = blockIdx.x % kchunks;
  int nt    = blockIdx.x / kchunks;
  int col   = nt * 16 + (lane & 15);
  int kbase = c * 32 + (lane >> 4) * 8;
  unsigned int* d = dst + ((size_t)(nt * kchunks + c) * 32 + lane) * 8;
#pragma unroll
  for (int j = 0; j < 8; ++j) {
    int k0 = (j < 4) ? (kbase + 2 * j) : (kbase + 16 + 2 * (j - 4)), k1 = k0 + 1;
    float v0 = (k0 < Cin) ? W[(size_t)k0 * Cout + col] : 0.f;
    float v1 = (k1 < Cin) ? W[(size_t)k1 * Cout + col] : 0.f;
    d[j] = pack_bf2(v0, v1);
  }
}

__global__ void zero_kernel(float* __restrict__ p, int n) {
  int i = blockIdx.x * blockDim.x + threadIdx.x;
  if (i < n) p[i] = 0.f;
}

// ---------------------------------------------------------------------------
// 4) WMMA GEMM: block = 8 waves = 8 consecutive 16-row tiles. Each wave loads
//    its A fragments ONCE into registers (KCH * 8 VGPRs) and sweeps all cout
//    tiles, reloading only the tiny L2-resident B fragments. Epilogue adds
//    bias, streams pre-act f32 with NT stores, and folds per-(batch,channel)
//    sum/sumsq via LDS atomics + global atomics.
// ---------------------------------------------------------------------------
template <int KCH>
__global__ void wmma_gemm_kernel(const unsigned int* __restrict__ Afrag,
                                 const unsigned int* __restrict__ Bfrag,
                                 const float* __restrict__ bias,
                                 float* __restrict__ y,
                                 float* __restrict__ s1, float* __restrict__ s2,
                                 int couttiles, int rowsPerB) {
  __shared__ float sh1[96], sh2[96];
  int lane = threadIdx.x & 31;
  int wave = threadIdx.x >> 5;
  int t    = blockIdx.x * 8 + wave;
  int Cout = couttiles * 16;
  for (int i = threadIdx.x; i < 96; i += blockDim.x) { sh1[i] = 0.f; sh2[i] = 0.f; }
  __syncthreads();

  // A fragments resident in registers for the whole cout sweep
  FragAB fa[KCH];
  {
    const unsigned int* ap = Afrag + (((size_t)t * KCH) * 32 + lane) * 8;
#pragma unroll
    for (int c = 0; c < KCH; ++c) {
      fa[c].q[0] = ((const uint4*)ap)[0];
      fa[c].q[1] = ((const uint4*)ap)[1];
      ap += 256;
    }
  }

  int mbase = t * 16 + ((lane >> 4) * 8);
  for (int nt = 0; nt < couttiles; ++nt) {
    v8f acc = {};
    const unsigned int* bp = Bfrag + (((size_t)nt * KCH) * 32 + lane) * 8;
#pragma unroll
    for (int c = 0; c < KCH; ++c) {
      FragAB fb;
      fb.q[0] = ((const uint4*)bp)[0];
      fb.q[1] = ((const uint4*)bp)[1];
      acc = __builtin_amdgcn_wmma_f32_16x16x32_bf16(false, fa[c].v, false, fb.v,
                                                    (short)0, acc, false, false);
      bp += 256;
    }
    int   col = nt * 16 + (lane & 15);
    float bv  = bias[col];
    float a1 = 0.f, a2 = 0.f;
#pragma unroll
    for (int r = 0; r < 8; ++r) {
      float v = acc[r] + bv;
      __builtin_nontemporal_store(v, &y[(size_t)(mbase + r) * Cout + col]);
      a1 += v; a2 += v * v;
    }
    if (s1) {
      atomicAdd(&sh1[col], a1);
      atomicAdd(&sh2[col], a2);
    }
  }

  if (s1) {
    __syncthreads();
    int b = (blockIdx.x * 128) / rowsPerB;   // 128 rows per block, batch-aligned
    for (int i = threadIdx.x; i < Cout; i += blockDim.x) {
      atomicAdd(&s1[b * Cout + i], sh1[i]);
      atomicAdd(&s2[b * Cout + i], sh2[i]);
    }
  }
}

// per (batch, group) mean / rsqrt(var+eps)
__global__ void gn_stats_kernel(const float* __restrict__ s1, const float* __restrict__ s2,
                                float* __restrict__ mu, float* __restrict__ rs,
                                int Cout, int rowsPerB) {
  int tid = threadIdx.x;                 // 64 = BATCH*8
  if (tid >= BATCH * 8) return;
  int b = tid >> 3, g = tid & 7;
  int cpg = Cout / 8;
  float sum = 0.f, sq = 0.f;
  for (int i = 0; i < cpg; ++i) {
    sum += s1[b * Cout + g * cpg + i];
    sq  += s2[b * Cout + g * cpg + i];
  }
  float count = (float)rowsPerB * (float)cpg;
  float m   = sum / count;
  float var = sq / count - m * m;
  mu[tid] = m;
  rs[tid] = rsqrtf(var + 1e-5f);
}

// GN + ReLU + bf16 repack into A-fragments for next layer (y streamed NT)
__global__ void repack_gn_kernel(const float* __restrict__ y, int C,
                                 const float* __restrict__ mu, const float* __restrict__ rs,
                                 const float* __restrict__ gamma, const float* __restrict__ beta,
                                 unsigned int* __restrict__ dst,
                                 int kchunks, int rowsPerB) {
  int tid  = blockIdx.x * blockDim.x + threadIdx.x;   // tiles*kchunks*32 threads
  int lane = tid & 31;
  int rest = tid >> 5;
  int c    = rest % kchunks;
  int t    = rest / kchunks;
  int row  = t * 16 + (lane & 15);
  int b    = row / rowsPerB;
  int khalf = (lane >> 4) * 8;
  int cpg   = C / 8;
  unsigned int* d = dst + ((size_t)(t * kchunks + c) * 32 + lane) * 8;
#pragma unroll
  for (int j = 0; j < 8; ++j) {
    int kk0 = c * 32 + a_pattern(khalf, j);
    float v0 = 0.f, v1 = 0.f;
    if (kk0 < C) {
      int g = kk0 / cpg;
      float x = __builtin_nontemporal_load(&y[(size_t)row * C + kk0]);
      v0 = fmaxf((x - mu[b * 8 + g]) * rs[b * 8 + g] * gamma[kk0] + beta[kk0], 0.f);
    }
    int kk1 = kk0 + 1;
    if (kk1 < C) {
      int g = kk1 / cpg;
      float x = __builtin_nontemporal_load(&y[(size_t)row * C + kk1]);
      v1 = fmaxf((x - mu[b * 8 + g]) * rs[b * 8 + g] * gamma[kk1] + beta[kk1], 0.f);
    }
    d[j] = pack_bf2(v0, v1);
  }
}

// raw bf16 repack (for pooled features, already activated)
__global__ void pack_raw_kernel(const float* __restrict__ x, int C,
                                unsigned int* __restrict__ dst, int kchunks) {
  int tid  = blockIdx.x * blockDim.x + threadIdx.x;
  int lane = tid & 31;
  int rest = tid >> 5;
  int c    = rest % kchunks;
  int t    = rest / kchunks;
  int row  = t * 16 + (lane & 15);
  int khalf = (lane >> 4) * 8;
  unsigned int* d = dst + ((size_t)(t * kchunks + c) * 32 + lane) * 8;
#pragma unroll
  for (int j = 0; j < 8; ++j) {
    int kk0 = c * 32 + a_pattern(khalf, j);
    float v0 = (kk0     < C) ? x[(size_t)row * C + kk0]     : 0.f;
    float v1 = (kk0 + 1 < C) ? x[(size_t)row * C + kk0 + 1] : 0.f;
    d[j] = pack_bf2(v0, v1);
  }
}

// layer-2 epilogue: GN + ReLU fused with max over K neighbors (y streamed NT)
__global__ void gn_maxpool_kernel(const float* __restrict__ y,
                                  const float* __restrict__ mu, const float* __restrict__ rs,
                                  const float* __restrict__ gamma, const float* __restrict__ beta,
                                  float* __restrict__ pooled) {
  int tid = blockIdx.x * blockDim.x + threadIdx.x;    // M_PT*96 threads
  int ch = tid % 96;
  int bn = tid / 96;
  int b  = bn / NPTS;
  int g  = ch / 12;
  float mm = mu[b * 8 + g], rr = rs[b * 8 + g], ga = gamma[ch], be = beta[ch];
  const float* base = y + (size_t)bn * KNBR * 96 + ch;
  float mx = 0.f;                                     // relu -> values >= 0
  for (int k = 0; k < KNBR; ++k) {
    float x = __builtin_nontemporal_load(&base[(size_t)k * 96]);
    float v = fmaxf((x - mm) * rr * ga + be, 0.f);
    mx = fmaxf(mx, v);
  }
  pooled[(size_t)bn * 96 + ch] = mx;
}

// final row L2 normalization -> output
__global__ void final_norm_kernel(const float* __restrict__ y, float* __restrict__ out) {
  int row = blockIdx.x * blockDim.x + threadIdx.x;    // M_PT threads
  const float* p = y + (size_t)row * 96;
  float ss = 0.f;
  for (int c = 0; c < 96; ++c) { float v = p[c]; ss += v * v; }
  float inv = rsqrtf(ss);
  float* o = out + (size_t)row * 96;
  for (int c = 0; c < 96; ++c) o[c] = p[c] * inv;
}

// ---------------------------------------------------------------------------
// Host launch sequence (graph-capture safe; scratch carved from d_ws).
// ---------------------------------------------------------------------------
extern "C" void kernel_launch(void* const* d_in, const int* in_sizes, int n_in,
                              void* d_out, int out_size, void* d_ws, size_t ws_size,
                              hipStream_t stream) {
  (void)in_sizes; (void)n_in; (void)out_size; (void)ws_size;
  const float* xyz = (const float*)d_in[0];
  const float* nrm = (const float*)d_in[1];
  const float* W0  = (const float*)d_in[2];   const float* b0  = (const float*)d_in[3];
  const float* g0  = (const float*)d_in[4];   const float* be0 = (const float*)d_in[5];
  const float* W1  = (const float*)d_in[6];   const float* b1  = (const float*)d_in[7];
  const float* g1  = (const float*)d_in[8];   const float* be1 = (const float*)d_in[9];
  const float* W2  = (const float*)d_in[10];  const float* b2  = (const float*)d_in[11];
  const float* g2  = (const float*)d_in[12];  const float* be2 = (const float*)d_in[13];
  const float* P0  = (const float*)d_in[14];  const float* pb0 = (const float*)d_in[15];
  const float* pg0 = (const float*)d_in[16];  const float* pbe0= (const float*)d_in[17];
  const float* P1  = (const float*)d_in[18];  const float* pb1 = (const float*)d_in[19];
  const float* pg1 = (const float*)d_in[20];  const float* pbe1= (const float*)d_in[21];
  const float* P2  = (const float*)d_in[22];  const float* pb2 = (const float*)d_in[23];
  float* out = (float*)d_out;

  char* ws = (char*)d_ws;
  size_t off = 0;
  auto take = [&](size_t bytes) -> char* {
    char* p = ws + off;
    off += (bytes + 255) & ~(size_t)255;
    return p;
  };

  int*          gidx   = (int*)         take((size_t)M_CONV * sizeof(int));
  unsigned int* fragA  = (unsigned int*)take((size_t)TILES_CONV * 2 * 256 * sizeof(unsigned int)); // 128 MB
  unsigned int* fragB  = (unsigned int*)take((size_t)TILES_CONV * 2 * 256 * sizeof(unsigned int)); // 128 MB
  float*        ybuf   = (float*)       take((size_t)M_CONV * 96 * sizeof(float));                 // 403 MB
  float*        pooled = (float*)       take((size_t)M_PT * 96 * sizeof(float));
  unsigned int* wf[6];
  for (int i = 0; i < 6; ++i) wf[i] = (unsigned int*)take((size_t)6 * 3 * 256 * sizeof(unsigned int));
  float* s1 = (float*)take((size_t)2 * BATCH * 96 * sizeof(float));
  float* s2 = s1 + BATCH * 96;
  float* mu = (float*)take((size_t)2 * BATCH * 8 * sizeof(float));
  float* rs = mu + BATCH * 8;

  const int RPB_CONV = NPTS * KNBR;   // group-norm rows per batch (conv stage)
  const int RPB_PT   = NPTS;          // (pointwise stage)

  // neighbor search + feature build
  knn_kernel <<<(BATCH * NPTS) / 4, 128, 0, stream>>>(xyz, gidx);
  feat_kernel<<<(TILES_CONV * 32) / 256, 256, 0, stream>>>(xyz, nrm, gidx, fragA);

  // weight pre-pack
  pack_weight_kernel<<<3 * 1, 32, 0, stream>>>(W0, 10, 48, wf[0], 1);
  pack_weight_kernel<<<3 * 2, 32, 0, stream>>>(W1, 48, 48, wf[1], 2);
  pack_weight_kernel<<<6 * 2, 32, 0, stream>>>(W2, 48, 96, wf[2], 2);
  pack_weight_kernel<<<6 * 3, 32, 0, stream>>>(P0, 96, 96, wf[3], 3);
  pack_weight_kernel<<<6 * 3, 32, 0, stream>>>(P1, 96, 96, wf[4], 3);
  pack_weight_kernel<<<6 * 3, 32, 0, stream>>>(P2, 96, 96, wf[5], 3);

  // ---- conv layer 0: 10 -> 48 ----
  zero_kernel<<<(2 * BATCH * 96 + 255) / 256, 256, 0, stream>>>(s1, 2 * BATCH * 96);
  wmma_gemm_kernel<1><<<TILES_CONV / 8, 256, 0, stream>>>(fragA, wf[0], b0, ybuf, s1, s2, 3, RPB_CONV);
  gn_stats_kernel<<<1, 64, 0, stream>>>(s1, s2, mu, rs, 48, RPB_CONV);
  repack_gn_kernel<<<(TILES_CONV * 2 * 32) / 256, 256, 0, stream>>>(ybuf, 48, mu, rs, g0, be0, fragB, 2, RPB_CONV);

  // ---- conv layer 1: 48 -> 48 ----
  zero_kernel<<<(2 * BATCH * 96 + 255) / 256, 256, 0, stream>>>(s1, 2 * BATCH * 96);
  wmma_gemm_kernel<2><<<TILES_CONV / 8, 256, 0, stream>>>(fragB, wf[1], b1, ybuf, s1, s2, 3, RPB_CONV);
  gn_stats_kernel<<<1, 64, 0, stream>>>(s1, s2, mu, rs, 48, RPB_CONV);
  repack_gn_kernel<<<(TILES_CONV * 2 * 32) / 256, 256, 0, stream>>>(ybuf, 48, mu, rs, g1, be1, fragA, 2, RPB_CONV);

  // ---- conv layer 2: 48 -> 96, then GN+ReLU+max over K ----
  zero_kernel<<<(2 * BATCH * 96 + 255) / 256, 256, 0, stream>>>(s1, 2 * BATCH * 96);
  wmma_gemm_kernel<2><<<TILES_CONV / 8, 256, 0, stream>>>(fragA, wf[2], b2, ybuf, s1, s2, 6, RPB_CONV);
  gn_stats_kernel<<<1, 64, 0, stream>>>(s1, s2, mu, rs, 96, RPB_CONV);
  gn_maxpool_kernel<<<(M_PT * 96) / 256, 256, 0, stream>>>(ybuf, mu, rs, g2, be2, pooled);
  pack_raw_kernel<<<(TILES_PT * 3 * 32) / 256, 256, 0, stream>>>(pooled, 96, fragB, 3);

  // ---- pointwise P0: 96 -> 96 ----
  zero_kernel<<<(2 * BATCH * 96 + 255) / 256, 256, 0, stream>>>(s1, 2 * BATCH * 96);
  wmma_gemm_kernel<3><<<TILES_PT / 8, 256, 0, stream>>>(fragB, wf[3], pb0, ybuf, s1, s2, 6, RPB_PT);
  gn_stats_kernel<<<1, 64, 0, stream>>>(s1, s2, mu, rs, 96, RPB_PT);
  repack_gn_kernel<<<(TILES_PT * 3 * 32) / 256, 256, 0, stream>>>(ybuf, 96, mu, rs, pg0, pbe0, fragA, 3, RPB_PT);

  // ---- pointwise P1: 96 -> 96 ----
  zero_kernel<<<(2 * BATCH * 96 + 255) / 256, 256, 0, stream>>>(s1, 2 * BATCH * 96);
  wmma_gemm_kernel<3><<<TILES_PT / 8, 256, 0, stream>>>(fragA, wf[4], pb1, ybuf, s1, s2, 6, RPB_PT);
  gn_stats_kernel<<<1, 64, 0, stream>>>(s1, s2, mu, rs, 96, RPB_PT);
  repack_gn_kernel<<<(TILES_PT * 3 * 32) / 256, 256, 0, stream>>>(ybuf, 96, mu, rs, pg1, pbe1, fragB, 3, RPB_PT);

  // ---- pointwise P2 (no GN) + L2 normalize ----
  wmma_gemm_kernel<3><<<TILES_PT / 8, 256, 0, stream>>>(fragB, wf[5], pb2, ybuf, nullptr, nullptr, 6, RPB_PT);
  final_norm_kernel<<<M_PT / 256, 256, 0, stream>>>(ybuf, out);
}
